// SeqParallelAttention_7885559955819
// MI455X (gfx1250) — matrix operand order repaired and checked by reading, not compiled, not run on hardware
//
#include <hip/hip_runtime.h>
#include <hip/hip_bf16.h>

// ---------------------------------------------------------------------------
// Problem constants (from reference): G=4, B=1, S=1024, D=2048, H=32, DH=64
// ---------------------------------------------------------------------------
#define ROWS   4096
#define DDIM   2048
#define NHEAD  32
#define SEQ    1024
#define DH     64
#define NKEYS  4096   // G*S

typedef __bf16 v16bf __attribute__((ext_vector_type(16)));
typedef __bf16 v8bf  __attribute__((ext_vector_type(8)));
typedef __bf16 v4bf  __attribute__((ext_vector_type(4)));
typedef float  v8f   __attribute__((ext_vector_type(8)));
typedef float  v4f   __attribute__((ext_vector_type(4)));

#define WMMA_BF16(a, b, c) \
  __builtin_amdgcn_wmma_f32_16x16x32_bf16(false, (a), false, (b), (short)0, (c), false, false)

// Build a 16-element bf16 fragment from two groups of 8 consecutive bf16.
static __device__ inline v16bf frag16(const __bf16* p, const __bf16* q) {
  v8bf a = *(const v8bf*)p;
  v8bf b = *(const v8bf*)q;
  return __builtin_shufflevector(a, b, 0,1,2,3,4,5,6,7,8,9,10,11,12,13,14,15);
}

// ---------------------------------------------------------------------------
// One-shot fp32 -> bf16 conversion (x and the four weight matrices), so the
// GEMM inner loops are pure global_load_b128 + v_wmma with no v_cvt traffic.
// ---------------------------------------------------------------------------
__global__ __launch_bounds__(256) void cvt_bf16(const float* __restrict__ in,
                                                __bf16* __restrict__ out, int n4) {
  int i = blockIdx.x * blockDim.x + threadIdx.x;
  const int stride = gridDim.x * blockDim.x;
  for (; i < n4; i += stride) {
    v4f x = ((const v4f*)in)[i];
    v4bf y;
#pragma unroll
    for (int e = 0; e < 4; ++e) y[e] = (__bf16)x[e];
    ((v4bf*)out)[i] = y;
  }
}

// ---------------------------------------------------------------------------
// GEMM: C[M,N] = A[M,K] @ W[N,K]^T + bias   (bf16 operands, fp32 accumulate)
// Block = 256 threads = 8 waves arranged 2 (M) x 4 (N); each wave owns 64x64.
// Workgroup tile: 128 x 256.  Grid = (N/256, M/128).
// A-fragment (16x32, MxK): lane(l&15)=M, K = e + 16*(l>>4).
// B-fragment (32x16, KxN): lane(l&15)=N (a W row), K = e + 16*(l>>4),
//   contiguous in memory since W is row-major [N,K] -> no transpose needed.
// C-fragment: VGPR r -> row r + 8*(l>>4), col = l&15.
// ---------------------------------------------------------------------------
template <typename TOUT>
__global__ __launch_bounds__(256) void gemm_xw_t(const __bf16* __restrict__ A,
                                                 const __bf16* __restrict__ W,
                                                 const float* __restrict__ bias,
                                                 TOUT* __restrict__ C,
                                                 int M, int N, int K) {
  const int lane = threadIdx.x & 31;
  const int wv   = threadIdx.x >> 5;
  const int wr   = wv >> 2;        // 0..1
  const int wc   = wv & 3;         // 0..3
  const int ml   = lane & 15;
  const int hi   = lane >> 4;
  const int row0 = blockIdx.y * 128 + wr * 64;
  const int col0 = blockIdx.x * 256 + wc * 64;

  v8f acc[4][4];
#pragma unroll
  for (int i = 0; i < 4; ++i)
#pragma unroll
    for (int j = 0; j < 4; ++j)
#pragma unroll
      for (int r = 0; r < 8; ++r) acc[i][j][r] = 0.f;

  for (int k0 = 0; k0 < K; k0 += 32) {
    v16bf af[4], bf[4];
#pragma unroll
    for (int i = 0; i < 4; ++i) {
      const __bf16* p = A + (size_t)(row0 + 16 * i + ml) * K + k0 + 16 * hi;
      af[i] = frag16(p, p + 8);
    }
#pragma unroll
    for (int j = 0; j < 4; ++j) {
      const __bf16* p = W + (size_t)(col0 + 16 * j + ml) * K + k0 + 16 * hi;
      bf[j] = frag16(p, p + 8);
    }
    if (k0 + 32 < K) {  // hint the next A tile (global_prefetch_b8)
      __builtin_prefetch(A + (size_t)(row0 + ml) * K + k0 + 32, 0, 0);
    }
#pragma unroll
    for (int i = 0; i < 4; ++i)
#pragma unroll
      for (int j = 0; j < 4; ++j)
        acc[i][j] = WMMA_BF16(af[i], bf[j], acc[i][j]);
  }

#pragma unroll
  for (int i = 0; i < 4; ++i)
#pragma unroll
    for (int j = 0; j < 4; ++j) {
      const int col = col0 + 16 * j + ml;
      const float bb = bias[col];
#pragma unroll
      for (int r = 0; r < 8; ++r) {
        const int row = row0 + 16 * i + r + 8 * hi;
        C[(size_t)row * N + col] = (TOUT)(acc[i][j][r] + bb);
      }
    }
}

// ---------------------------------------------------------------------------
// RMSNorm (over full D=2048, before head split) + repack into attention layouts:
//   Qbf: [g][h][s][d]       (queries; 1/sqrt(DH) score scale folded in here)
//   Kbf: [h][j = g*S+s][d]  (keys, row-major per head, ld=64)
//   Vt : [h][d][j]          (values TRANSPOSED so PV B-fragments are contiguous)
// ---------------------------------------------------------------------------
__global__ __launch_bounds__(256) void rms_pack(const __bf16* __restrict__ qpre,
                                                const __bf16* __restrict__ kpre,
                                                const __bf16* __restrict__ vpre,
                                                const float* __restrict__ qn_w,
                                                const float* __restrict__ kn_w,
                                                __bf16* __restrict__ Qbf,
                                                __bf16* __restrict__ Kbf,
                                                __bf16* __restrict__ Vt) {
  const int row = blockIdx.x;            // 0..4095 = g*S + s
  const int g = row >> 10, s = row & 1023;
  const int t = threadIdx.x;
  __shared__ float red[256];

  float qv[8], kv[8];
  float ssq = 0.f, ssk = 0.f;
#pragma unroll
  for (int i = 0; i < 8; ++i) {
    const int col = t * 8 + i;
    const float q = (float)qpre[(size_t)row * DDIM + col];
    const float k = (float)kpre[(size_t)row * DDIM + col];
    qv[i] = q; kv[i] = k;
    ssq += q * q; ssk += k * k;
  }
  red[t] = ssq; __syncthreads();
  for (int off = 128; off > 0; off >>= 1) { if (t < off) red[t] += red[t + off]; __syncthreads(); }
  const float rq = rsqrtf(red[0] * (1.f / DDIM) + 1e-6f) * 0.125f;  // * 1/sqrt(DH)
  __syncthreads();
  red[t] = ssk; __syncthreads();
  for (int off = 128; off > 0; off >>= 1) { if (t < off) red[t] += red[t + off]; __syncthreads(); }
  const float rk = rsqrtf(red[0] * (1.f / DDIM) + 1e-6f);

#pragma unroll
  for (int i = 0; i < 8; ++i) {
    const int col = t * 8 + i;
    const int h = col >> 6, d = col & 63;
    Qbf[((size_t)(g * NHEAD + h) * SEQ + s) * DH + d] = (__bf16)(qv[i] * rq * qn_w[col]);
    Kbf[((size_t)h * NKEYS + row) * DH + d]           = (__bf16)(kv[i] * rk * kn_w[col]);
    Vt [((size_t)h * DH + d) * NKEYS + row]           = vpre[(size_t)row * DDIM + col];
  }
}

// ---------------------------------------------------------------------------
// Flash attention, one (g,h) per blockIdx.y, 8 waves/block, each wave owns 16
// query rows with an independent online softmax over all 4096 keys.
// Key tile = 128: the cross-lane max reduction (batched shfl_xor steps so the
// 8 ds_bpermute per step issue under one wait), the accumulator rescale, and
// the m-update exp all amortize over 128 keys. Row-SUM is folded into WMMA
// (lacc += P @ ones) on the bf16 P fragments the PV product needs anyway.
// ---------------------------------------------------------------------------
__global__ __launch_bounds__(256) void attn_kernel(const __bf16* __restrict__ Qbf,
                                                   const __bf16* __restrict__ Kbf,
                                                   const __bf16* __restrict__ Vt,
                                                   __bf16* __restrict__ O /* [4096, 2048] */) {
  const int lane = threadIdx.x & 31;
  const int wv   = threadIdx.x >> 5;
  const int ml   = lane & 15;
  const int hi   = lane >> 4;
  const int gh   = blockIdx.y;             // 0..127
  const int g = gh >> 5, h = gh & 31;
  const int q0 = blockIdx.x * 128 + wv * 16;

  __shared__ __bf16 plds[8][16 * 128];     // per-wave P staging: 4KB x 8 = 32KB
  __bf16* pl = plds[wv];

  const __bf16* Qh = Qbf + (size_t)gh * SEQ * DH;
  const __bf16* Kh = Kbf + (size_t)h * NKEYS * DH;
  const __bf16* Vh = Vt  + (size_t)h * DH * NKEYS;

  // Persistent Q fragments: 16 rows x 64 (two 16x32 A-frags, pre-scaled)
  v16bf qf[2];
#pragma unroll
  for (int kk = 0; kk < 2; ++kk) {
    const __bf16* p = Qh + (size_t)(q0 + ml) * DH + kk * 32 + 16 * hi;
    qf[kk] = frag16(p, p + 8);
  }

  v16bf ones;
#pragma unroll
  for (int e = 0; e < 16; ++e) ones[e] = (__bf16)1.f;

  v8f out[4], lacc;                        // lacc = running softmax denominator
#pragma unroll
  for (int dt = 0; dt < 4; ++dt)
#pragma unroll
    for (int r = 0; r < 8; ++r) out[dt][r] = 0.f;
#pragma unroll
  for (int r = 0; r < 8; ++r) lacc[r] = 0.f;

  float m[8];
#pragma unroll
  for (int r = 0; r < 8; ++r) m[r] = -1e30f;

  for (int j0 = 0; j0 < NKEYS; j0 += 128) {
    // ---- scores: 8 16x16 key tiles, K-dim = 64 (2 WMMA steps each) ----
    v8f sc[8];
#pragma unroll
    for (int nt = 0; nt < 8; ++nt)
#pragma unroll
      for (int r = 0; r < 8; ++r) sc[nt][r] = 0.f;
#pragma unroll
    for (int nt = 0; nt < 8; ++nt) {
#pragma unroll
      for (int kk = 0; kk < 2; ++kk) {
        const __bf16* p = Kh + (size_t)(j0 + nt * 16 + ml) * DH + kk * 32 + 16 * hi;
        sc[nt] = WMMA_BF16(qf[kk], frag16(p, p + 8), sc[nt]);
      }
    }
    // ---- per-row local max over the 8 tiles (pure VALU, no shuffles) ----
    float tmax[8];
#pragma unroll
    for (int r = 0; r < 8; ++r) {
      float t = fmaxf(sc[0][r], sc[1][r]);
      t = fmaxf(t, fmaxf(sc[2][r], sc[3][r]));
      t = fmaxf(t, fmaxf(sc[4][r], sc[5][r]));
      t = fmaxf(t, fmaxf(sc[6][r], sc[7][r]));
      tmax[r] = t;
    }
    // ---- cross-lane max: step-outer so 8 bpermutes issue per wait ----
#pragma unroll
    for (int step = 1; step <= 8; step <<= 1) {
      float o[8];
#pragma unroll
      for (int r = 0; r < 8; ++r) o[r] = __shfl_xor(tmax[r], step, 32);
#pragma unroll
      for (int r = 0; r < 8; ++r) tmax[r] = fmaxf(tmax[r], o[r]);
    }
    // ---- online update; stage P (bf16) to LDS ----
#pragma unroll
    for (int r = 0; r < 8; ++r) {
      const float mn = fmaxf(m[r], tmax[r]);
      const float alpha = __expf(m[r] - mn);
      m[r] = mn;
      lacc[r] *= alpha;
#pragma unroll
      for (int dt = 0; dt < 4; ++dt) out[dt][r] *= alpha;
      const int rowp = r + 8 * hi;         // actual P row of this lane/reg
#pragma unroll
      for (int nt = 0; nt < 8; ++nt)
        pl[rowp * 128 + nt * 16 + ml] = (__bf16)__expf(sc[nt][r] - mn);
    }
    asm volatile("s_wait_dscnt 0" ::: "memory");   // P stores -> P frag loads
    // ---- P is 16x128 = 4 A-frags; per frag: lacc WMMA + 4 PV WMMAs ----
#pragma unroll
    for (int kt = 0; kt < 4; ++kt) {
      v16bf pf;
#pragma unroll
      for (int e = 0; e < 16; ++e) {
        const int k = kt * 32 + 16 * hi + e;
        pf[e] = pl[ml * 128 + k];
      }
      lacc = WMMA_BF16(pf, ones, lacc);    // row sums of this P slice
#pragma unroll
      for (int dt = 0; dt < 4; ++dt) {
        const __bf16* p = Vh + (size_t)(dt * 16 + ml) * NKEYS + j0 + kt * 32 + 16 * hi;
        out[dt] = WMMA_BF16(pf, frag16(p, p + 8), out[dt]);
      }
    }
  }

  // ---- epilogue: normalize and write back to [g*S+s, h*64+d] (bf16) ----
#pragma unroll
  for (int dt = 0; dt < 4; ++dt) {
#pragma unroll
    for (int r = 0; r < 8; ++r) {
      const float o = out[dt][r] / lacc[r];
      const int row = g * SEQ + q0 + r + 8 * hi;
      const int col = h * DH + dt * 16 + ml;
      O[(size_t)row * DDIM + col] = (__bf16)o;
    }
  }
}

// ---------------------------------------------------------------------------
// Host-side orchestration. Workspace layout (MB):
//   [  0) xbf   16 | [ 16) Wq/Wk/Wv/Wo bf16 4x8 = 32
//   [ 48) qpre  16   (reused as attention output after rms_pack)
//   [ 64) kpre  16 | [ 80) vpre 16
//   [ 96) Qbf   16 | [112) Kbf  16 | [128) Vt 16   -> total 144MB
// ---------------------------------------------------------------------------
extern "C" void kernel_launch(void* const* d_in, const int* in_sizes, int n_in,
                              void* d_out, int out_size, void* d_ws, size_t ws_size,
                              hipStream_t stream) {
  const float* x  = (const float*)d_in[0];
  const float* Wq = (const float*)d_in[1];
  const float* bq = (const float*)d_in[2];
  const float* Wk = (const float*)d_in[3];
  const float* bk = (const float*)d_in[4];
  const float* Wv = (const float*)d_in[5];
  const float* bv = (const float*)d_in[6];
  const float* qn = (const float*)d_in[7];
  const float* kn = (const float*)d_in[8];
  const float* Wo = (const float*)d_in[9];
  const float* bo = (const float*)d_in[10];

  char* ws = (char*)d_ws;
  const size_t MB = 1u << 20;
  __bf16* xbf  = (__bf16*)(ws +   0 * MB);
  __bf16* Wqb  = (__bf16*)(ws +  16 * MB);
  __bf16* Wkb  = (__bf16*)(ws +  24 * MB);
  __bf16* Wvb  = (__bf16*)(ws +  32 * MB);
  __bf16* Wob  = (__bf16*)(ws +  40 * MB);
  __bf16* qpre = (__bf16*)(ws +  48 * MB);
  __bf16* kpre = (__bf16*)(ws +  64 * MB);
  __bf16* vpre = (__bf16*)(ws +  80 * MB);
  __bf16* Qbf  = (__bf16*)(ws +  96 * MB);
  __bf16* Kbf  = (__bf16*)(ws + 112 * MB);
  __bf16* Vt   = (__bf16*)(ws + 128 * MB);
  __bf16* attn_out = qpre;   // safe reuse: rms_pack finished reading qpre

  const dim3 blk(256);
  const dim3 gcv(2048);
  cvt_bf16<<<gcv, blk, 0, stream>>>(x,  xbf, ROWS * DDIM / 4);
  cvt_bf16<<<gcv, blk, 0, stream>>>(Wq, Wqb, DDIM * DDIM / 4);
  cvt_bf16<<<gcv, blk, 0, stream>>>(Wk, Wkb, DDIM * DDIM / 4);
  cvt_bf16<<<gcv, blk, 0, stream>>>(Wv, Wvb, DDIM * DDIM / 4);
  cvt_bf16<<<gcv, blk, 0, stream>>>(Wo, Wob, DDIM * DDIM / 4);

  const dim3 gproj(DDIM / 256, ROWS / 128);   // (8, 32)
  gemm_xw_t<__bf16><<<gproj, blk, 0, stream>>>(xbf, Wqb, bq, qpre, ROWS, DDIM, DDIM);
  gemm_xw_t<__bf16><<<gproj, blk, 0, stream>>>(xbf, Wkb, bk, kpre, ROWS, DDIM, DDIM);
  gemm_xw_t<__bf16><<<gproj, blk, 0, stream>>>(xbf, Wvb, bv, vpre, ROWS, DDIM, DDIM);

  rms_pack<<<dim3(ROWS), blk, 0, stream>>>(qpre, kpre, vpre, qn, kn, Qbf, Kbf, Vt);

  attn_kernel<<<dim3(SEQ / 128, 4 * NHEAD), blk, 0, stream>>>(Qbf, Kbf, Vt, attn_out);

  gemm_xw_t<float><<<gproj, blk, 0, stream>>>(attn_out, Wob, bo, (float*)d_out,
                                              ROWS, DDIM, DDIM);
}
